// GraphSAGEModel_24257975287900
// MI455X (gfx1250) — matrix operand order repaired and checked by reading, not compiled, not run on hardware
//
#include <hip/hip_runtime.h>

// GraphSAGE link prediction, MI455X (gfx1250, wave32).
// f32 master data, f16 WMMA (v_wmma_f32_16x16x32_f16) for all GEMMs,
// f32 global atomics for mean aggregation (L2-resident working set).
// Weights staged into LDS pre-swizzled into the WMMA B-fragment layout so each
// fragment fetch is one 32-byte LDS vector load (2x ds_load_b128).
// Accumulators start from inline C=0 (v_wmma ..., 0); bias folded into epilogue.

typedef _Float16 v16h __attribute__((ext_vector_type(16)));
typedef _Float16 v8h  __attribute__((ext_vector_type(8)));
typedef float    v8f  __attribute__((ext_vector_type(8)));
typedef float    v4f  __attribute__((ext_vector_type(4)));

#define HID 128

// ---- WMMA fragment layouts (CDNA5 ISA 7.12.2, 16-bit) ----
// A (16x32 MxK): lane L -> row M=L%16; halves 0..7 -> K=koff..koff+7,
//                halves 8..15 -> K=koff+16..koff+23, koff = (L>=16 ? 8 : 0).
// B (32x16 KxN): lane L -> col N=L%16; same K order as A (koff, koff+16 split).
// C/D f32: VGPR r -> M = r + (lane>=16 ? 8 : 0), N = lane%16.

__device__ __forceinline__ v8f wmma16(v16h a, v16h b, v8f c) {
  return __builtin_amdgcn_wmma_f32_16x16x32_f16(
      /*neg_a=*/false, a, /*neg_b=*/false, b,
      /*c_mod=*/(short)0, c, /*reuse_a=*/false, /*reuse_b=*/false);
}

// Stage a 128x128 f32 weight matrix (k-major: W[k*128+n]) into LDS in
// B-fragment-swizzled f16 layout: chunk t = j*128 + kb*32 + lane holds the 16
// contiguous halves lane `lane` needs for column-tile j, k-block kb.
__device__ __forceinline__ void stage_w_swz(const float* __restrict__ W,
                                            _Float16* __restrict__ sWb,
                                            int tid, int nthreads) {
  for (int t = tid; t < 1024; t += nthreads) {
    const int j = t >> 7;
    const int rem = t & 127;
    const int kb = rem >> 5;
    const int l = rem & 31;
    const int n = j * 16 + (l & 15);
    const int kstart = kb * 32 + ((l >= 16) ? 8 : 0);
    v16h v;
#pragma unroll
    for (int i = 0; i < 16; ++i) v[i] = (_Float16)W[(kstart + (i < 8 ? i : i + 8)) * HID + n];
    *(v16h*)(sWb + t * 16) = v;  // 32B contiguous -> 2x ds_store_b128
  }
}

// Build an A fragment from a global f32 row with 4x 16B vector loads.
__device__ __forceinline__ v16h make_afrag(const float* __restrict__ row,
                                           int k0, int off8, float scale) {
  const v4f q0 = *(const v4f*)(row + k0 + off8);
  const v4f q1 = *(const v4f*)(row + k0 + off8 + 4);
  const v4f q2 = *(const v4f*)(row + k0 + off8 + 16);
  const v4f q3 = *(const v4f*)(row + k0 + off8 + 20);
  v16h a;
#pragma unroll
  for (int i = 0; i < 4; ++i) {
    a[i]      = (_Float16)(q0[i] * scale);
    a[4 + i]  = (_Float16)(q1[i] * scale);
    a[8 + i]  = (_Float16)(q2[i] * scale);
    a[12 + i] = (_Float16)(q3[i] * scale);
  }
  return a;
}

// A fragment of elementwise product of two global f32 rows.
__device__ __forceinline__ v16h make_zfrag(const float* __restrict__ rs,
                                           const float* __restrict__ rd,
                                           int k0, int off8) {
  const v4f s0 = *(const v4f*)(rs + k0 + off8);
  const v4f s1 = *(const v4f*)(rs + k0 + off8 + 4);
  const v4f s2 = *(const v4f*)(rs + k0 + off8 + 16);
  const v4f s3 = *(const v4f*)(rs + k0 + off8 + 20);
  const v4f d0 = *(const v4f*)(rd + k0 + off8);
  const v4f d1 = *(const v4f*)(rd + k0 + off8 + 4);
  const v4f d2 = *(const v4f*)(rd + k0 + off8 + 16);
  const v4f d3 = *(const v4f*)(rd + k0 + off8 + 20);
  v16h a;
#pragma unroll
  for (int i = 0; i < 4; ++i) {
    a[i]      = (_Float16)(s0[i] * d0[i]);
    a[4 + i]  = (_Float16)(s1[i] * d1[i]);
    a[8 + i]  = (_Float16)(s2[i] * d2[i]);
    a[12 + i] = (_Float16)(s3[i] * d3[i]);
  }
  return a;
}

// ----------------------------- utility kernels -----------------------------
__global__ void zero_f32(float* __restrict__ p, long long n) {
  long long i = (long long)blockIdx.x * blockDim.x + threadIdx.x;
  const long long stride = (long long)gridDim.x * blockDim.x;
  for (; i < n; i += stride) p[i] = 0.0f;
}

// one thread per (edge, 4 features): agg[dst][f..f+3] += h[src][f..f+3]
__global__ void edge_scatter(const float* __restrict__ h,
                             const int* __restrict__ esrc,
                             const int* __restrict__ edst,
                             float* __restrict__ agg,
                             float* __restrict__ deg,
                             long long total4, int with_deg) {
  long long i = (long long)blockIdx.x * blockDim.x + threadIdx.x;
  const long long stride = (long long)gridDim.x * blockDim.x;
  for (; i < total4; i += stride) {
    const int e = (int)(i >> 5);
    const int f = (int)(i & 31) << 2;
    const int s = esrc[e];
    const int d = edst[e];
    const v4f val = *(const v4f*)(h + (long long)s * HID + f);
    float* dstp = agg + (long long)d * HID + f;
    atomicAdd(dstp + 0, val[0]);
    atomicAdd(dstp + 1, val[1]);
    atomicAdd(dstp + 2, val[2]);
    atomicAdd(dstp + 3, val[3]);
    if (with_deg && f == 0) atomicAdd(&deg[d], 1.0f);
    const long long nx = i + stride;
    if (nx < total4) __builtin_prefetch(&esrc[(int)(nx >> 5)], 0, 0);
  }
}

// ------------------- fused SAGEConv GEMM (WMMA, f16 in LDS) ----------------
// out[m][:] = act( A[m]@Wself + (agg[m]*invdeg[m])@Wneigh + bias )
// 256 threads = 8 waves; wave w owns rows [blk*128 + 16w, +16); full 128 cols.
template <bool RELU>
__global__ void __launch_bounds__(256)
sage_gemm(const float* __restrict__ A, const float* __restrict__ agg,
          const float* __restrict__ deg, const float* __restrict__ Wself,
          const float* __restrict__ Wneigh, const float* __restrict__ bias,
          float* __restrict__ out, int nrows) {
  __shared__ __align__(32) _Float16 sWs[HID * HID];  // 32 KB, swizzled
  __shared__ __align__(32) _Float16 sWn[HID * HID];  // 32 KB, swizzled

  const int tid = threadIdx.x;
  stage_w_swz(Wself, sWs, tid, 256);
  stage_w_swz(Wneigh, sWn, tid, 256);
  __syncthreads();

  const int wave = tid >> 5;
  const int lane = tid & 31;
  const int off8 = (lane >= 16) ? 8 : 0;
  const int mbase = blockIdx.x * 128 + wave * 16;
  const int row = mbase + (lane & 15);
  const int rowc = row < nrows ? row : (nrows - 1);

  const float invd = 1.0f / fmaxf(deg[rowc], 1.0f);
  const float* rowA = A + (long long)rowc * HID;
  const float* rowG = agg + (long long)rowc * HID;

  // Preload all A fragments (self + mean) for K = 0..127 (4 k-blocks).
  v16h aS[4], aM[4];
#pragma unroll
  for (int kb = 0; kb < 4; ++kb) {
    aS[kb] = make_afrag(rowA, kb * 32, off8, 1.0f);
    aM[kb] = make_afrag(rowG, kb * 32, off8, invd);
  }

  const v16h* sWsv = (const v16h*)sWs;
  const v16h* sWnv = (const v16h*)sWn;

#pragma unroll
  for (int j = 0; j < 8; ++j) {  // 8 column tiles of 16
    const int n = j * 16 + (lane & 15);
    const float bv = bias[n];
    v8f acc = {};  // inline SRC2 = 0 for the first WMMA

#pragma unroll
    for (int kb = 0; kb < 4; ++kb) {
      const int cidx = j * 128 + kb * 32 + lane;  // swizzled chunk index
      acc = wmma16(aS[kb], sWsv[cidx], acc);
      acc = wmma16(aM[kb], sWnv[cidx], acc);
    }

#pragma unroll
    for (int r = 0; r < 8; ++r) {
      const int m = mbase + r + ((lane >= 16) ? 8 : 0);
      if (m < nrows) {
        float v = acc[r] + bv;  // bias in epilogue
        if (RELU) v = fmaxf(v, 0.0f);
        out[(long long)m * HID + n] = v;
      }
    }
  }
}

// --------------------- fused edge decoder (3-layer MLP) --------------------
// 128 threads = 4 waves; wave w owns 16 pairs. Phased weight staging keeps
// static LDS at 32 KB (weights) + 16 KB (per-wave z scratch).
__global__ void __launch_bounds__(128)
edge_decode(const float* __restrict__ h2, const int* __restrict__ src,
            const int* __restrict__ dst, const float* __restrict__ Wd1,
            const float* __restrict__ bd1, const float* __restrict__ Wd2,
            const float* __restrict__ bd2, const float* __restrict__ Wd3,
            const float* __restrict__ bd3, float* __restrict__ out, int P) {
  __shared__ __align__(32) _Float16 sW[HID * HID];    // 32 KB, swizzled
  __shared__ __align__(32) _Float16 sZ[4][16 * HID];  // 16 KB, [m][k] f16

  const int tid = threadIdx.x;
  const int wave = tid >> 5;
  const int lane = tid & 31;
  const int off8 = (lane >= 16) ? 8 : 0;
  const int pbase = (blockIdx.x * 4 + wave) * 16;
  const int pair = pbase + (lane & 15);
  const int pc = pair < P ? pair : (P - 1);
  const int s = src[pc];
  const int d = dst[pc];
  const v16h* sWv = (const v16h*)sW;

  // ---- phase 1: z1 = relu((h[s]*h[d]) @ Wd1 + bd1) ----
  stage_w_swz(Wd1, sW, tid, 128);
  __syncthreads();

  const float* rs = h2 + (long long)s * HID;
  const float* rd = h2 + (long long)d * HID;
  v16h aZ[4];
#pragma unroll
  for (int kb = 0; kb < 4; ++kb) aZ[kb] = make_zfrag(rs, rd, kb * 32, off8);

#pragma unroll
  for (int j = 0; j < 8; ++j) {
    const int n = j * 16 + (lane & 15);
    const float bv = bd1[n];
    v8f acc = {};
#pragma unroll
    for (int kb = 0; kb < 4; ++kb)
      acc = wmma16(aZ[kb], sWv[j * 128 + kb * 32 + lane], acc);
#pragma unroll
    for (int r = 0; r < 8; ++r) {
      const int m = r + ((lane >= 16) ? 8 : 0);
      sZ[wave][m * HID + n] = (_Float16)fmaxf(acc[r] + bv, 0.0f);
    }
  }
  __syncthreads();

  // ---- phase 2: z2 = relu(z1 @ Wd2 + bd2) ----
  stage_w_swz(Wd2, sW, tid, 128);
  __syncthreads();

  const int mloc = lane & 15;
#pragma unroll
  for (int kb = 0; kb < 4; ++kb) {
    const _Float16* zr = &sZ[wave][mloc * HID + kb * 32 + off8];
    const v8h lo = *(const v8h*)zr;         // K = koff .. koff+7
    const v8h hi = *(const v8h*)(zr + 16);  // K = koff+16 .. koff+23
    v16h a;
#pragma unroll
    for (int i = 0; i < 8; ++i) { a[i] = lo[i]; a[8 + i] = hi[i]; }
    aZ[kb] = a;
  }
  __syncthreads();  // everyone done reading sZ before overwrite

#pragma unroll
  for (int j = 0; j < 8; ++j) {
    const int n = j * 16 + (lane & 15);
    const float bv = bd2[n];
    v8f acc = {};
#pragma unroll
    for (int kb = 0; kb < 4; ++kb)
      acc = wmma16(aZ[kb], sWv[j * 128 + kb * 32 + lane], acc);
#pragma unroll
    for (int r = 0; r < 8; ++r) {
      const int m = r + ((lane >= 16) ? 8 : 0);
      sZ[wave][m * HID + n] = (_Float16)fmaxf(acc[r] + bv, 0.0f);
    }
  }
  __syncthreads();

  // ---- phase 3: out = z2 @ Wd3 + bd3 (128 -> 1) ----
  if (lane < 16) {
    float sum = bd3[0];
#pragma unroll
    for (int c = 0; c < 16; ++c) {
      const v8h zq = *(const v8h*)&sZ[wave][lane * HID + c * 8];
      const v4f w0 = *(const v4f*)(Wd3 + c * 8);
      const v4f w1 = *(const v4f*)(Wd3 + c * 8 + 4);
#pragma unroll
      for (int i = 0; i < 4; ++i)
        sum += (float)zq[i] * w0[i] + (float)zq[4 + i] * w1[i];
    }
    const int pm = pbase + lane;
    if (pm < P) out[pm] = sum;
  }
}

// ------------------------------- launcher ----------------------------------
extern "C" void kernel_launch(void* const* d_in, const int* in_sizes, int n_in,
                              void* d_out, int out_size, void* d_ws,
                              size_t ws_size, hipStream_t stream) {
  const int N = in_sizes[0] / HID;
  const int E = in_sizes[1];
  const int P = in_sizes[3];

  const float* x    = (const float*)d_in[0];
  const int*   esrc = (const int*)d_in[1];
  const int*   edst = (const int*)d_in[2];
  const int*   psrc = (const int*)d_in[3];
  const int*   pdst = (const int*)d_in[4];
  const int*   nsrc = (const int*)d_in[5];
  const int*   ndst = (const int*)d_in[6];
  const float* Ws1  = (const float*)d_in[7];
  const float* Wn1  = (const float*)d_in[8];
  const float* b1   = (const float*)d_in[9];
  const float* Ws2  = (const float*)d_in[10];
  const float* Wn2  = (const float*)d_in[11];
  const float* b2   = (const float*)d_in[12];
  const float* Wd1  = (const float*)d_in[13];
  const float* bd1  = (const float*)d_in[14];
  const float* Wd2  = (const float*)d_in[15];
  const float* bd2  = (const float*)d_in[16];
  const float* Wd3  = (const float*)d_in[17];
  const float* bd3  = (const float*)d_in[18];
  float* out = (float*)d_out;

  char* w = (char*)d_ws;
  auto take = [&](size_t bytes) {
    void* p = (void*)w;
    w += (bytes + 255) & ~(size_t)255;
    return p;
  };
  float* deg = (float*)take((size_t)N * sizeof(float));
  float* agg = (float*)take((size_t)N * HID * sizeof(float));
  float* h1  = (float*)take((size_t)N * HID * sizeof(float));
  float* h2  = (float*)take((size_t)N * HID * sizeof(float));

  const long long eTot4 = (long long)E * (HID / 4);
  const int sBlocks = 4096;
  const int gemmBlocks = (N + 127) / 128;
  const int decBlocks = (P + 63) / 64;

  // layer 1
  zero_f32<<<2048, 256, 0, stream>>>(deg, N);
  zero_f32<<<2048, 256, 0, stream>>>(agg, (long long)N * HID);
  edge_scatter<<<sBlocks, 256, 0, stream>>>(x, esrc, edst, agg, deg, eTot4, 1);
  sage_gemm<true><<<gemmBlocks, 256, 0, stream>>>(x, agg, deg, Ws1, Wn1, b1,
                                                  h1, N);
  // layer 2
  zero_f32<<<2048, 256, 0, stream>>>(agg, (long long)N * HID);
  edge_scatter<<<sBlocks, 256, 0, stream>>>(h1, esrc, edst, agg, deg, eTot4, 0);
  sage_gemm<false><<<gemmBlocks, 256, 0, stream>>>(h1, agg, deg, Ws2, Wn2, b2,
                                                   h2, N);
  // decoder: pos then neg, concatenated in d_out
  edge_decode<<<decBlocks, 128, 0, stream>>>(h2, psrc, pdst, Wd1, bd1, Wd2,
                                             bd2, Wd3, bd3, out, P);
  edge_decode<<<decBlocks, 128, 0, stream>>>(h2, nsrc, ndst, Wd1, bd1, Wd2,
                                             bd2, Wd3, bd3, out + P, P);
}
